// LocalNeighborhoodAttention_7730941133357
// MI455X (gfx1250) — matrix operand order repaired and checked by reading, not compiled, not run on hardware
//
#include <hip/hip_runtime.h>
#include <hip/hip_bf16.h>

// ---------------------------------------------------------------------------
// LocalNeighborhoodAttention, MI455X (gfx1250, wave32, WMMA + async-to-LDS)
//
//   knn  = top16-smallest(distance row)                   (kernel 1)
//   H16 = f16(H); Wt*16 = f16(W*)^T                       (kernel 0, once)
//   Qf=H@Wq, Kf=H@Wk, Vf=H@Wv                             (kernel 2, WMMA f16,
//                                                          A and B both staged by
//                                                          global_load_async_to_lds)
//   logits=Qf[n].Kf[knn]*scale; softmax; Eh16=attn@Vf     (kernel 3, L2 gathers)
//   out = Eh@Wo + bo + H                                  (kernel 4, WMMA f16)
// ---------------------------------------------------------------------------

typedef __attribute__((ext_vector_type(16))) _Float16 v16h;
typedef __attribute__((ext_vector_type(8)))  _Float16 v8h;
typedef __attribute__((ext_vector_type(4)))  _Float16 v4h;
typedef __attribute__((ext_vector_type(8)))  float    v8f;

constexpr int NN  = 8192;   // nodes
constexpr int CC  = 512;    // channels == HID
constexpr int KNB = 16;     // neighbors
constexpr float ATT_SCALE = 0.04419417382415922f; // 512^-0.5

// --------------------------- f32 -> f16 convert ----------------------------
__global__ __launch_bounds__(256) void cvt_f16_kernel(
    const float* __restrict__ src, _Float16* __restrict__ dst, int n4)
{
  const int i = blockIdx.x * 256 + threadIdx.x;
  if (i < n4) {
    const float4 v = reinterpret_cast<const float4*>(src)[i];
    v4h h = { (_Float16)v.x, (_Float16)v.y, (_Float16)v.z, (_Float16)v.w };
    reinterpret_cast<v4h*>(dst)[i] = h;
  }
}

// ---------------- f32 -> f16 + transpose (weights, one-time) ---------------
// Wt[n][k] = (f16) W[k][n]; 32x32 LDS tiles so both sides stay coalesced.
__global__ __launch_bounds__(256) void cvt_transpose_f16_kernel(
    const float* __restrict__ W, _Float16* __restrict__ Wt)
{
  __shared__ float t[32][33];
  const int tx = threadIdx.x;             // 0..31
  const int ty = threadIdx.y;             // 0..7
  const int k0 = blockIdx.x * 32;
  const int n0 = blockIdx.y * 32;
  #pragma unroll
  for (int j = 0; j < 4; ++j)
    t[ty + j * 8][tx] = W[(size_t)(k0 + ty + j * 8) * CC + n0 + tx];
  __syncthreads();
  #pragma unroll
  for (int j = 0; j < 4; ++j)
    Wt[(size_t)(n0 + ty + j * 8) * CC + k0 + tx] = (_Float16)t[tx][ty + j * 8];
}

// ------------------------------- top-k -------------------------------------
__global__ __launch_bounds__(256) void topk16_kernel(
    const float* __restrict__ dist, int* __restrict__ knn)
{
  __shared__ float sdist[NN];            // 32 KB row cache
  __shared__ float wmin[8];
  __shared__ int   widx[8];

  const int row  = blockIdx.x;
  const int tid  = threadIdx.x;
  const int lane = tid & 31;
  const int wave = tid >> 5;
  const float* drow = dist + (size_t)row * NN;

  for (int i = tid; i < NN; i += 256) sdist[i] = drow[i];
  __syncthreads();

  for (int sel = 0; sel < KNB; ++sel) {
    float best = __builtin_inff();
    int   bidx = NN;
    for (int i = tid; i < NN; i += 256) {
      const float v = sdist[i];
      if (v < best || (v == best && i < bidx)) { best = v; bidx = i; }
    }
    #pragma unroll
    for (int off = 16; off > 0; off >>= 1) {   // wave32 reduction
      const float ob = __shfl_xor(best, off, 32);
      const int   oi = __shfl_xor(bidx, off, 32);
      if (ob < best || (ob == best && oi < bidx)) { best = ob; bidx = oi; }
    }
    if (lane == 0) { wmin[wave] = best; widx[wave] = bidx; }
    __syncthreads();
    if (tid == 0) {
      float b = wmin[0]; int bi = widx[0];
      #pragma unroll
      for (int w = 1; w < 8; ++w)
        if (wmin[w] < b || (wmin[w] == b && widx[w] < bi)) { b = wmin[w]; bi = widx[w]; }
      knn[row * KNB + sel] = bi;
      sdist[bi] = __builtin_inff();
    }
    __syncthreads();
  }
}

// ------------------------------- WMMA GEMM ---------------------------------
// C(8192x512 f32) = A(8192x512 f16, row-major) @ B (given as Bt = B^T, f16).
// Block tile 128x64, 8 waves, wave tile 32x32 (2x2 of v_wmma_f32_16x16x32_f16).
// Both operands double-buffered in LDS, filled only by
// global_load_async_to_lds_b128 (ASYNCcnt) -- no VGPR transit, no loadcnt
// stalls in the compute loop.
constexpr int BM = 128, BN = 64, BK = 32;
constexpr int LDS_STRIDE = BK + 8;       // 40 halves = 80B (multiple of 16B)

__device__ __forceinline__ v16h join16(v8h lo, v8h hi) {
  return __builtin_shufflevector(lo, hi, 0,1,2,3,4,5,6,7,8,9,10,11,12,13,14,15);
}

__device__ __forceinline__ void wait_async0() {
  asm volatile("s_wait_asynccnt 0x0" ::: "memory");
}

__device__ __forceinline__ void async_cp16(const _Float16* gsrc, _Float16* ldst) {
  const unsigned int       la = (unsigned int)(uintptr_t)ldst;
  const unsigned long long ga = (unsigned long long)(const void*)gsrc;
  asm volatile("global_load_async_to_lds_b128 %0, %1, off"
               :: "v"(la), "v"(ga) : "memory");
}

// A tile BMxBK: 512 x 16B chunks, 2 per lane.
__device__ __forceinline__ void stage_A_async(
    const _Float16* __restrict__ A, _Float16* AsBuf, int tileM, int k0, int tid)
{
  #pragma unroll
  for (int i = 0; i < 2; ++i) {
    const int c   = tid + i * 256;     // 0..511
    const int row = c >> 2;            // 0..127
    const int seg = (c & 3) * 8;       // halves 0,8,16,24
    async_cp16(A + (size_t)(tileM + row) * CC + k0 + seg,
               AsBuf + row * LDS_STRIDE + seg);
  }
}

// B tile (pre-transposed [n][k]) BNxBK: 256 x 16B chunks, 1 per lane.
__device__ __forceinline__ void stage_B_async(
    const _Float16* __restrict__ Bt, _Float16* BsBuf, int tileN, int k0, int tid)
{
  const int row = tid >> 2;            // 0..63
  const int seg = (tid & 3) * 8;       // halves 0,8,16,24
  async_cp16(Bt + (size_t)(tileN + row) * CC + k0 + seg,
             BsBuf + row * LDS_STRIDE + seg);
}

__device__ __forceinline__ void gemm_tile_core(
    const _Float16* __restrict__ A, const _Float16* __restrict__ Bt,
    float* __restrict__ C, const float* __restrict__ bias,
    const float* __restrict__ residual)
{
  __shared__ alignas(16) _Float16 As[2][BM * LDS_STRIDE];  // 2 x 10240 B
  __shared__ alignas(16) _Float16 Bs[2][BN * LDS_STRIDE];  // 2 x  5120 B

  const int tid   = threadIdx.x;
  const int lane  = tid & 31;
  const int wave  = tid >> 5;
  const int g     = lane >> 4;          // lane group (ISA 16x16 layout)
  const int lm    = lane & 15;
  const int tileM = blockIdx.y * BM;
  const int tileN = blockIdx.x * BN;
  const int waveM = (wave >> 1) * 32;   // 4x2 wave grid
  const int waveN = (wave & 1) * 32;

  v8f acc00 = {}, acc01 = {}, acc10 = {}, acc11 = {};

  // prologue: DMA buffer 0
  stage_A_async(A,  As[0], tileM, 0, tid);
  stage_B_async(Bt, Bs[0], tileN, 0, tid);
  wait_async0();
  __syncthreads();

  int buf = 0;
  for (int k0 = 0; k0 < CC; k0 += BK) {
    // overlap: DMA next k-slice into the other buffer while computing
    if (k0 + BK < CC) {
      stage_A_async(A,  As[buf ^ 1], tileM, k0 + BK, tid);
      stage_B_async(Bt, Bs[buf ^ 1], tileN, k0 + BK, tid);
    }

    // fragments (ISA 7.12.2 16-bit layouts)
    // A 16x32: lane=M(mod16); halves 0..7 -> K=g*8.., halves 8..15 -> +16.
    v16h a0, a1, b0, b1;
    {
      const _Float16* p = &As[buf][(waveM + lm) * LDS_STRIDE + g * 8];
      a0 = join16(*(const v8h*)p, *(const v8h*)(p + 16));
      p += 16 * LDS_STRIDE;
      a1 = join16(*(const v8h*)p, *(const v8h*)(p + 16));
    }
    // B 32x16: lane=N(mod16); halves 0..7 -> K=g*8.., halves 8..15 -> +16.
    {
      const _Float16* p = &Bs[buf][(waveN + lm) * LDS_STRIDE + g * 8];
      b0 = join16(*(const v8h*)p, *(const v8h*)(p + 16));
      p += 16 * LDS_STRIDE;
      b1 = join16(*(const v8h*)p, *(const v8h*)(p + 16));
    }

    acc00 = __builtin_amdgcn_wmma_f32_16x16x32_f16(false, a0, false, b0,
                                                   (short)0, acc00, false, false);
    acc01 = __builtin_amdgcn_wmma_f32_16x16x32_f16(false, a0, false, b1,
                                                   (short)0, acc01, false, false);
    acc10 = __builtin_amdgcn_wmma_f32_16x16x32_f16(false, a1, false, b0,
                                                   (short)0, acc10, false, false);
    acc11 = __builtin_amdgcn_wmma_f32_16x16x32_f16(false, a1, false, b1,
                                                   (short)0, acc11, false, false);

    wait_async0();       // next-buffer DMA landed
    __syncthreads();     // all waves done with this phase
    buf ^= 1;
  }

  // epilogue: VGPR v -> M = v (lanes 0-15) / 8+v (lanes 16-31), N = lane%16
  auto store_tile = [&](const v8f& acc, int mBase, int nBase) {
    const int col = tileN + nBase + lm;
    #pragma unroll
    for (int v = 0; v < 8; ++v) {
      const int row = tileM + mBase + (g ? 8 + v : v);
      float val = acc[v];
      if (bias)     val += bias[col];
      if (residual) val += residual[(size_t)row * CC + col];
      C[(size_t)row * CC + col] = val;
    }
  };
  store_tile(acc00, waveM +  0, waveN +  0);
  store_tile(acc01, waveM +  0, waveN + 16);
  store_tile(acc10, waveM + 16, waveN +  0);
  store_tile(acc11, waveM + 16, waveN + 16);
}

__global__ __launch_bounds__(256) void gemm_qkv_kernel(
    const _Float16* __restrict__ H16,
    const _Float16* __restrict__ Wqt, const _Float16* __restrict__ Wkt,
    const _Float16* __restrict__ Wvt,
    float* __restrict__ Qf, float* __restrict__ Kf, float* __restrict__ Vf)
{
  const _Float16* Bt; float* C;
  if      (blockIdx.z == 0) { Bt = Wqt; C = Qf; }
  else if (blockIdx.z == 1) { Bt = Wkt; C = Kf; }
  else                      { Bt = Wvt; C = Vf; }
  gemm_tile_core(H16, Bt, C, nullptr, nullptr);
}

__global__ __launch_bounds__(256) void gemm_out_kernel(
    const _Float16* __restrict__ Eh16, const _Float16* __restrict__ Wot,
    const float* __restrict__ bo, const float* __restrict__ H,
    float* __restrict__ out)
{
  gemm_tile_core(Eh16, Wot, out, bo, H);
}

// --------------------- gather + softmax + weighted sum ---------------------
__global__ __launch_bounds__(256) void attn_kernel(
    const float* __restrict__ Qf, const float* __restrict__ Kf,
    const float* __restrict__ Vf, const int* __restrict__ knn,
    _Float16* __restrict__ Eh16)
{
  __shared__ int   sidx[KNB];
  __shared__ float slog[KNB];
  __shared__ float qs[CC];

  const int n    = blockIdx.x;
  const int tid  = threadIdx.x;
  const int lane = tid & 31;
  const int wave = tid >> 5;

  if (tid < KNB) sidx[tid] = knn[n * KNB + tid];
  qs[tid]       = Qf[(size_t)n * CC + tid];
  qs[tid + 256] = Qf[(size_t)n * CC + tid + 256];
  __syncthreads();

  // wave w computes logits for neighbors 2w, 2w+1 (512-dim dots, L2-resident)
  #pragma unroll
  for (int j = 0; j < 2; ++j) {
    const int k = wave * 2 + j;
    const float* kp = Kf + (size_t)sidx[k] * CC;
    float p = 0.f;
    #pragma unroll
    for (int c = 0; c < CC / 32; ++c) p += qs[lane + c * 32] * kp[lane + c * 32];
    #pragma unroll
    for (int off = 16; off > 0; off >>= 1) p += __shfl_xor(p, off, 32);
    if (lane == 0) slog[k] = p * ATT_SCALE;
  }
  __syncthreads();

  float m = -__builtin_inff();
  #pragma unroll
  for (int k = 0; k < KNB; ++k) m = fmaxf(m, slog[k]);
  float w[KNB], den = 0.f;
  #pragma unroll
  for (int k = 0; k < KNB; ++k) { w[k] = __expf(slog[k] - m); den += w[k]; }
  const float inv = 1.f / den;

  #pragma unroll
  for (int cc = 0; cc < 2; ++cc) {
    const int c = tid + cc * 256;
    float acc = 0.f;
    #pragma unroll
    for (int k = 0; k < KNB; ++k)
      acc += w[k] * Vf[(size_t)sidx[k] * CC + c];
    Eh16[(size_t)n * CC + c] = (_Float16)(acc * inv);   // f16 for final WMMA
  }
}

// ------------------------------- launch ------------------------------------
extern "C" void kernel_launch(void* const* d_in, const int* in_sizes, int n_in,
                              void* d_out, int out_size, void* d_ws, size_t ws_size,
                              hipStream_t stream)
{
  const float* H  = (const float*)d_in[0];
  const float* Dm = (const float*)d_in[1];
  const float* Wq = (const float*)d_in[2];
  const float* Wk = (const float*)d_in[3];
  const float* Wv = (const float*)d_in[4];
  const float* Wo = (const float*)d_in[5];
  const float* bo = (const float*)d_in[6];
  float* out = (float*)d_out;

  const size_t MB = 1ull << 20;
  char* ws = (char*)d_ws;
  int*      knn  = (int*)ws;                              // 512 KB
  float*    Qf   = (float*)   (ws +  1 * MB);             // 16 MB
  float*    Kf   = (float*)   (ws + 17 * MB);             // 16 MB
  float*    Vf   = (float*)   (ws + 33 * MB);             // 16 MB
  _Float16* H16  = (_Float16*)(ws + 49 * MB);             //  8 MB
  _Float16* Wqt  = (_Float16*)(ws + 57 * MB);             // 512 KB each (B^T)
  _Float16* Wkt  = Wqt + (size_t)CC * CC;
  _Float16* Wvt  = Wkt + (size_t)CC * CC;
  _Float16* Wot  = Wvt + (size_t)CC * CC;
  _Float16* Eh16 = (_Float16*)(ws + 60 * MB);             //  8 MB

  // one-time operand prep: H -> f16; weights -> f16 transposed
  cvt_f16_kernel<<<(NN * CC / 4 + 255) / 256, 256, 0, stream>>>(H, H16, NN * CC / 4);
  {
    dim3 tb(32, 8), tg(CC / 32, CC / 32);
    cvt_transpose_f16_kernel<<<tg, tb, 0, stream>>>(Wq, Wqt);
    cvt_transpose_f16_kernel<<<tg, tb, 0, stream>>>(Wk, Wkt);
    cvt_transpose_f16_kernel<<<tg, tb, 0, stream>>>(Wv, Wvt);
    cvt_transpose_f16_kernel<<<tg, tb, 0, stream>>>(Wo, Wot);
  }

  topk16_kernel<<<NN, 256, 0, stream>>>(Dm, knn);

  dim3 gqkv(CC / BN, NN / BM, 3);
  gemm_qkv_kernel<<<gqkv, 256, 0, stream>>>(H16, Wqt, Wkt, Wvt, Qf, Kf, Vf);

  attn_kernel<<<NN, 256, 0, stream>>>(Qf, Kf, Vf, knn, Eh16);

  dim3 gout(CC / BN, NN / BM, 1);
  gemm_out_kernel<<<gout, 256, 0, stream>>>(Eh16, Wot, bo, H, out);
}
